// GPT_40123584479789
// MI455X (gfx1250) — matrix operand order repaired and checked by reading, not compiled, not run on hardware
//
#include <hip/hip_runtime.h>
#include <hip/hip_bf16.h>

// GPT-2 small forward on gfx1250 via v_wmma_f32_16x16x32_f16.
// Double-buffered LDS staging, async-to-LDS when the toolchain exposes it.
// Workspace carved below needs ~625 MB.

typedef _Float16 v16h __attribute__((ext_vector_type(16)));
typedef float    v8f  __attribute__((ext_vector_type(8)));
typedef int      gv4i __attribute__((vector_size(16)));

#define GPT_V  50257
#define GPT_L  12
#define GPT_H  12
#define GPT_D  768
#define GPT_HD 64
#define GPT_DF 3072
#define GPT_B  4
#define GPT_T  1024
#define GPT_BT (GPT_B * GPT_T)

#if defined(__HIP_DEVICE_COMPILE__) && \
    __has_builtin(__builtin_amdgcn_global_load_async_to_lds_b128) && \
    __has_builtin(__builtin_amdgcn_s_wait_asynccnt)
#define GPT_ASYNC 1
#else
#define GPT_ASYNC 0
#endif

#if GPT_ASYNC
__device__ __forceinline__ void async_b128(const _Float16* g, _Float16* l) {
  __builtin_amdgcn_global_load_async_to_lds_b128(
      (__attribute__((address_space(1))) gv4i*)g,
      (__attribute__((address_space(3))) gv4i*)l,
      0, 0);
}
#endif

// ---------------- reductions (wave32) ----------------
__device__ inline float warp_sum(float v) {
  #pragma unroll
  for (int o = 16; o; o >>= 1) v += __shfl_xor(v, o, 32);
  return v;
}
__device__ inline float warp_max(float v) {
  #pragma unroll
  for (int o = 16; o; o >>= 1) v = fmaxf(v, __shfl_xor(v, o, 32));
  return v;
}
__device__ float block_sum(float v, float* s) {
  v = warp_sum(v);
  int w = threadIdx.x >> 5, l = threadIdx.x & 31;
  if (l == 0) s[w] = v;
  __syncthreads();
  if (w == 0) {
    float t = (l < (int)(blockDim.x >> 5)) ? s[l] : 0.f;
    t = warp_sum(t);
    if (l == 0) s[0] = t;
  }
  __syncthreads();
  float r = s[0];
  __syncthreads();
  return r;
}
__device__ float block_max(float v, float* s) {
  v = warp_max(v);
  int w = threadIdx.x >> 5, l = threadIdx.x & 31;
  if (l == 0) s[w] = v;
  __syncthreads();
  if (w == 0) {
    float t = (l < (int)(blockDim.x >> 5)) ? s[l] : -3.4e38f;
    t = warp_max(t);
    if (l == 0) s[0] = t;
  }
  __syncthreads();
  float r = s[0];
  __syncthreads();
  return r;
}

__device__ inline float gelu_exact(float x) {
  return 0.5f * x * (1.f + erff(x * 0.70710678118654752f));
}

// ---------------- small kernels ----------------
__global__ __launch_bounds__(256)
void cvt_f32_f16_kernel(const float* __restrict__ src, _Float16* __restrict__ dst, long long n) {
  long long i = (long long)blockIdx.x * blockDim.x + threadIdx.x;
  long long stride = (long long)gridDim.x * blockDim.x;
  for (; i < n; i += stride) dst[i] = (_Float16)src[i];
}

__global__ __launch_bounds__(256)
void embed_kernel(const int* __restrict__ idx, const float* __restrict__ wte,
                  const float* __restrict__ wpe, float* __restrict__ x) {
  int t = blockIdx.x;                 // 0 .. B*T-1
  int tok = idx[t];
  int pos = t % GPT_T;
  const float* we = wte + (size_t)tok * GPT_D;
  const float* pe = wpe + (size_t)pos * GPT_D;
  float* xr = x + (size_t)t * GPT_D;
  for (int d = threadIdx.x; d < GPT_D; d += blockDim.x) xr[d] = we[d] + pe[d];
}

__global__ __launch_bounds__(256)
void ln_f16_kernel(const float* __restrict__ x, const float* __restrict__ g,
                   const float* __restrict__ b, _Float16* __restrict__ out) {
  __shared__ float s[32];
  int t = blockIdx.x;
  const float* row = x + (size_t)t * GPT_D;
  float lsum = 0.f, lsq = 0.f;
  for (int d = threadIdx.x; d < GPT_D; d += blockDim.x) {
    float v = row[d];
    lsum += v; lsq += v * v;
  }
  float mean = block_sum(lsum, s) * (1.f / GPT_D);
  float msq  = block_sum(lsq,  s) * (1.f / GPT_D);
  float rstd = rsqrtf(msq - mean * mean + 1e-5f);
  _Float16* orow = out + (size_t)t * GPT_D;
  for (int d = threadIdx.x; d < GPT_D; d += blockDim.x)
    orow[d] = (_Float16)((row[d] - mean) * rstd * g[d] + b[d]);
}

__global__ __launch_bounds__(256)
void softmax_causal_kernel(const float* __restrict__ scores, _Float16* __restrict__ probs) {
  __shared__ float s[32];
  int q = blockIdx.x;
  long long base = ((long long)blockIdx.y * GPT_T + q) * GPT_T;  // blockIdx.y = b*H+h
  const float* row = scores + base;
  _Float16* prow = probs + base;
  float lmax = -3.4e38f;
  for (int k = threadIdx.x; k < GPT_T; k += blockDim.x)
    if (k <= q) lmax = fmaxf(lmax, row[k]);
  float m = block_max(lmax, s);
  float lsum = 0.f;
  for (int k = threadIdx.x; k < GPT_T; k += blockDim.x)
    if (k <= q) lsum += __expf(row[k] - m);
  float inv = 1.f / block_sum(lsum, s);
  for (int k = threadIdx.x; k < GPT_T; k += blockDim.x)
    prow[k] = (k <= q) ? (_Float16)(__expf(row[k] - m) * inv) : (_Float16)0.f;
}

// V transpose: vT[(b*H+h)*HD + hd][t] = qkv[b*T+t][2D + h*HD + hd]
__global__ __launch_bounds__(64)
void transpose_v_kernel(const _Float16* __restrict__ qkv, _Float16* __restrict__ vT) {
  int t = blockIdx.x, bh = blockIdx.y;
  int b = bh / GPT_H, h = bh % GPT_H;
  const _Float16* src = qkv + ((size_t)(b * GPT_T + t)) * (3 * GPT_D) + 2 * GPT_D + h * GPT_HD;
  _Float16* dst = vT + ((size_t)bh * GPT_HD) * GPT_T + t;
  int d = threadIdx.x;
  dst[(size_t)d * GPT_T] = src[d];
}

// ---------------- WMMA GEMM ----------------
// C[M,N] = act( alpha * A[M,K] * W^T (+R) ),  W: [N,K] row-major, ldb = row stride.
// Batched via blockIdx.z: zb = z/batH, zh = z%batH, pointer offsets zb*Outer + zh*Inner.
// Block: 256 threads = 8 waves; tile 128(M) x 64(N); wave tile 32x32.
// Double-buffered LDS; async global->LDS DMA when available.
__global__ __launch_bounds__(256)
void wmma_gemm_kernel(const _Float16* __restrict__ A, long long aOuter, long long aInner, int lda,
                      const _Float16* __restrict__ Bm, long long bOuter, long long bInner, int ldb,
                      const float* R, float* Cf, _Float16* Ch,
                      long long cOuter, long long cInner, int ldc,
                      int M, int N, int K, int batH,
                      float alpha, int act, int causal) {
  __shared__ _Float16 As[2][128 * 40];   // 32 K-halves + 8 pad per row (80B rows, 16B aligned)
  __shared__ _Float16 Bs[2][64 * 40];

  int bm = blockIdx.y * 128, bn = blockIdx.x * 64;
  if (causal && bn > bm + 127) return;   // tile fully above the diagonal

  int zb = (int)blockIdx.z / batH, zh = (int)blockIdx.z % batH;
  A  += (long long)zb * aOuter + (long long)zh * aInner;
  Bm += (long long)zb * bOuter + (long long)zh * bInner;
  long long coff = (long long)zb * cOuter + (long long)zh * cInner;

  int tid = threadIdx.x;
  int lane = tid & 31, g = lane >> 4, r = lane & 15;
  int wv = tid >> 5, waveM = wv & 3, waveN = wv >> 2;   // 4 x 2 wave grid
  int brow = tid >> 2, bkc = tid & 3;                   // B staging coords

  v8f acc[2][2] = {};
  int nk = K >> 5;

#if GPT_ASYNC
  auto stage_async = [&](int buf, int k0) {
    #pragma unroll
    for (int rep = 0; rep < 2; ++rep) {
      int idx = tid + rep * 256;
      int row = idx >> 2, kc = idx & 3;
      async_b128(A + (size_t)(bm + row) * lda + k0 + kc * 8,
                 &As[buf][row * 40 + kc * 8]);
    }
    int gn = bn + brow;
    if (gn < N)   // OOB rows keep stale LDS; masked at the epilogue store
      async_b128(Bm + (size_t)gn * ldb + k0 + bkc * 8,
                 &Bs[buf][brow * 40 + bkc * 8]);
  };
  stage_async(0, 0);
#else
  uint4 ra[2], rb;
  auto stage_load = [&](int k0) {
    #pragma unroll
    for (int rep = 0; rep < 2; ++rep) {
      int idx = tid + rep * 256;
      int row = idx >> 2, kc = idx & 3;
      ra[rep] = *(reinterpret_cast<const uint4*>(A + (size_t)(bm + row) * lda + k0) + kc);
    }
    int gn = bn + brow;
    if (gn < N) {
      rb = *(reinterpret_cast<const uint4*>(Bm + (size_t)gn * ldb + k0) + bkc);
    } else {
      rb.x = rb.y = rb.z = rb.w = 0u;
    }
  };
  auto stage_store = [&](int buf) {
    #pragma unroll
    for (int rep = 0; rep < 2; ++rep) {
      int idx = tid + rep * 256;
      int row = idx >> 2, kc = idx & 3;
      *reinterpret_cast<uint4*>(&As[buf][row * 40 + kc * 8]) = ra[rep];
    }
    *reinterpret_cast<uint4*>(&Bs[buf][brow * 40 + bkc * 8]) = rb;
  };
  stage_load(0);
  stage_store(0);
#endif

  for (int i = 0; i < nk; ++i) {
    int cur = i & 1;
#if GPT_ASYNC
    __builtin_amdgcn_s_wait_asynccnt(0);
#endif
    __syncthreads();
    // kick off next slab so DMA/VMEM overlaps this slab's WMMA
#if GPT_ASYNC
    if (i + 1 < nk) stage_async(cur ^ 1, (i + 1) << 5);
#else
    if (i + 1 < nk) stage_load((i + 1) << 5);
#endif

    // fragments per ISA 7.12.2 layouts (all 16B-aligned ds_load_b128)
    union F { v16h v; uint4 q[2]; };
    F af[2], bf[2];
    #pragma unroll
    for (int mi = 0; mi < 2; ++mi) {
      const _Float16* p = &As[cur][(waveM * 32 + mi * 16 + r) * 40];
      af[mi].q[0] = *reinterpret_cast<const uint4*>(p + g * 8);
      af[mi].q[1] = *reinterpret_cast<const uint4*>(p + 16 + g * 8);
    }
    #pragma unroll
    for (int ni = 0; ni < 2; ++ni) {
      const _Float16* p = &Bs[cur][(waveN * 32 + ni * 16 + r) * 40];
      bf[ni].q[0] = *reinterpret_cast<const uint4*>(p + g * 8);
      bf[ni].q[1] = *reinterpret_cast<const uint4*>(p + 16 + g * 8);
    }

    #pragma unroll
    for (int mi = 0; mi < 2; ++mi)
      #pragma unroll
      for (int ni = 0; ni < 2; ++ni)
        acc[mi][ni] = __builtin_amdgcn_wmma_f32_16x16x32_f16(
            false, af[mi].v, false, bf[ni].v, (short)0, acc[mi][ni], false, false);

#if !GPT_ASYNC
    if (i + 1 < nk) stage_store(cur ^ 1);
    __syncthreads();
#endif
  }

  // epilogue: D layout M = v + 8*g, N = r
  const float* Rp  = R  ? R  + coff : nullptr;
  float*    Cfp    = Cf ? Cf + coff : nullptr;
  _Float16* Chp    = Ch ? Ch + coff : nullptr;
  #pragma unroll
  for (int mi = 0; mi < 2; ++mi) {
    #pragma unroll
    for (int ni = 0; ni < 2; ++ni) {
      #pragma unroll
      for (int v = 0; v < 8; ++v) {
        int gm = bm + waveM * 32 + mi * 16 + v + 8 * g;
        int gn = bn + waveN * 32 + ni * 16 + r;
        if (gm < M && gn < N) {
          float val = alpha * acc[mi][ni][v];
          long long cidx = (long long)gm * ldc + gn;
          if (Rp) val += Rp[cidx];
          if (act == 1) val = gelu_exact(val);
          if (Cfp) Cfp[cidx] = val;
          if (Chp) Chp[cidx] = (_Float16)val;
        }
      }
    }
  }
}

// ---------------- host launcher ----------------
extern "C" void kernel_launch(void* const* d_in, const int* in_sizes, int n_in,
                              void* d_out, int out_size, void* d_ws, size_t ws_size,
                              hipStream_t stream) {
  (void)in_sizes; (void)n_in; (void)out_size; (void)ws_size;
  const int*   idx   = (const int*)  d_in[0];
  const float* wte   = (const float*)d_in[1];
  const float* wpe   = (const float*)d_in[2];
  const float* ln1_g = (const float*)d_in[3];
  const float* ln1_b = (const float*)d_in[4];
  const float* qkv_w = (const float*)d_in[5];
  const float* out_w = (const float*)d_in[6];
  const float* ln2_g = (const float*)d_in[7];
  const float* ln2_b = (const float*)d_in[8];
  const float* fc1_w = (const float*)d_in[9];
  const float* fc2_w = (const float*)d_in[10];
  const float* lnf_g = (const float*)d_in[11];
  const float* lnf_b = (const float*)d_in[12];
  float* logits = (float*)d_out;

  char* wsp = (char*)d_ws;
  auto carve = [&](size_t bytes) -> char* {
    char* p = wsp;
    wsp += (bytes + 255) & ~(size_t)255;
    return p;
  };
  _Float16* wte16  = (_Float16*)carve((size_t)GPT_V * GPT_D * 2);
  _Float16* qkvw16 = (_Float16*)carve((size_t)GPT_L * 3 * GPT_D * GPT_D * 2);
  _Float16* ow16   = (_Float16*)carve((size_t)GPT_L * GPT_D * GPT_D * 2);
  _Float16* f1w16  = (_Float16*)carve((size_t)GPT_L * GPT_DF * GPT_D * 2);
  _Float16* f2w16  = (_Float16*)carve((size_t)GPT_L * GPT_D * GPT_DF * 2);
  float*    x      = (float*)   carve((size_t)GPT_BT * GPT_D * 4);
  _Float16* h16    = (_Float16*)carve((size_t)GPT_BT * GPT_D * 2);
  _Float16* qkv16  = (_Float16*)carve((size_t)GPT_BT * 3 * GPT_D * 2);
  float*    scores = (float*)   carve((size_t)GPT_B * GPT_H * GPT_T * GPT_T * 4);
  _Float16* probs  = (_Float16*)carve((size_t)GPT_B * GPT_H * GPT_T * GPT_T * 2);
  _Float16* o16    = (_Float16*)carve((size_t)GPT_BT * GPT_D * 2);
  _Float16* a16    = (_Float16*)carve((size_t)GPT_BT * GPT_DF * 2);
  _Float16* vT     = (_Float16*)carve((size_t)GPT_B * GPT_H * GPT_HD * GPT_T * 2);

  dim3 blk(256);
  auto cvt = [&](const float* s, _Float16* dptr, long long n) {
    long long grid = (n + 2047) / 2048;
    if (grid > 65535) grid = 65535;
    hipLaunchKernelGGL(cvt_f32_f16_kernel, dim3((unsigned)grid), blk, 0, stream, s, dptr, n);
  };
  cvt(wte,   wte16,  (long long)GPT_V * GPT_D);
  cvt(qkv_w, qkvw16, (long long)GPT_L * 3 * GPT_D * GPT_D);
  cvt(out_w, ow16,   (long long)GPT_L * GPT_D * GPT_D);
  cvt(fc1_w, f1w16,  (long long)GPT_L * GPT_DF * GPT_D);
  cvt(fc2_w, f2w16,  (long long)GPT_L * GPT_D * GPT_DF);

  hipLaunchKernelGGL(embed_kernel, dim3(GPT_BT), blk, 0, stream, idx, wte, wpe, x);

  auto gemm = [&](const _Float16* A, long long aO, long long aI, int lda,
                  const _Float16* Bp, long long bO, long long bI, int ldb,
                  const float* R, float* Cf, _Float16* Ch,
                  long long cO, long long cI, int ldc,
                  int M, int N, int K, int batches, int batH,
                  float alpha, int act, int causal) {
    dim3 grid((N + 63) / 64, M / 128, batches);
    hipLaunchKernelGGL(wmma_gemm_kernel, grid, blk, 0, stream,
                       A, aO, aI, lda, Bp, bO, bI, ldb, R, Cf, Ch,
                       cO, cI, ldc, M, N, K, batH, alpha, act, causal);
  };

  const long long sAq = (long long)GPT_T * 3 * GPT_D;   // per-batch stride inside qkv16

  for (int l = 0; l < GPT_L; ++l) {
    hipLaunchKernelGGL(ln_f16_kernel, dim3(GPT_BT), blk, 0, stream,
                       x, ln1_g + (size_t)l * GPT_D, ln1_b + (size_t)l * GPT_D, h16);
    // qkv[BT,3D] = h * Wqkv^T
    gemm(h16, 0, 0, GPT_D,
         qkvw16 + (size_t)l * 3 * GPT_D * GPT_D, 0, 0, GPT_D,
         nullptr, nullptr, qkv16, 0, 0, 3 * GPT_D,
         GPT_BT, 3 * GPT_D, GPT_D, 1, 1, 1.f, 0, 0);
    // scores[b,h] = Q K^T / sqrt(64); tiles above the diagonal skipped
    gemm(qkv16,          sAq, GPT_HD, 3 * GPT_D,
         qkv16 + GPT_D,  sAq, GPT_HD, 3 * GPT_D,
         nullptr, scores, nullptr,
         (long long)GPT_H * GPT_T * GPT_T, (long long)GPT_T * GPT_T, GPT_T,
         GPT_T, GPT_T, GPT_HD, GPT_B * GPT_H, GPT_H, 0.125f, 0, 1);
    hipLaunchKernelGGL(softmax_causal_kernel, dim3(GPT_T, GPT_B * GPT_H), blk, 0, stream,
                       scores, probs);
    hipLaunchKernelGGL(transpose_v_kernel, dim3(GPT_T, GPT_B * GPT_H), dim3(64), 0, stream,
                       qkv16, vT);
    // o[b,t,h,hd] = P V   (B operand = vT[bh][hd][t], [N,K] row-major)
    gemm(probs, (long long)GPT_H * GPT_T * GPT_T, (long long)GPT_T * GPT_T, GPT_T,
         vT, (long long)GPT_H * GPT_HD * GPT_T, (long long)GPT_HD * GPT_T, GPT_T,
         nullptr, nullptr, o16,
         (long long)GPT_T * GPT_D, GPT_HD, GPT_D,
         GPT_T, GPT_HD, GPT_T, GPT_B * GPT_H, GPT_H, 1.f, 0, 0);
    // x += o * Wout^T
    gemm(o16, 0, 0, GPT_D,
         ow16 + (size_t)l * GPT_D * GPT_D, 0, 0, GPT_D,
         x, x, nullptr, 0, 0, GPT_D,
         GPT_BT, GPT_D, GPT_D, 1, 1, 1.f, 0, 0);
    hipLaunchKernelGGL(ln_f16_kernel, dim3(GPT_BT), blk, 0, stream,
                       x, ln2_g + (size_t)l * GPT_D, ln2_b + (size_t)l * GPT_D, h16);
    // a = gelu(h * W1^T)
    gemm(h16, 0, 0, GPT_D,
         f1w16 + (size_t)l * GPT_DF * GPT_D, 0, 0, GPT_D,
         nullptr, nullptr, a16, 0, 0, GPT_DF,
         GPT_BT, GPT_DF, GPT_D, 1, 1, 1.f, 1, 0);
    // x += a * W2^T
    gemm(a16, 0, 0, GPT_DF,
         f2w16 + (size_t)l * GPT_D * GPT_DF, 0, 0, GPT_DF,
         x, x, nullptr, 0, 0, GPT_D,
         GPT_BT, GPT_D, GPT_DF, 1, 1, 1.f, 0, 0);
  }

  hipLaunchKernelGGL(ln_f16_kernel, dim3(GPT_BT), blk, 0, stream, x, lnf_g, lnf_b, h16);
  // logits = h * wte^T  (N = 50257, masked tail)
  gemm(h16, 0, 0, GPT_D,
       wte16, 0, 0, GPT_D,
       nullptr, logits, nullptr, 0, 0, GPT_V,
       GPT_BT, GPT_V, GPT_D, 1, 1, 1.f, 0, 0);
}
